// Module_75823352644386
// MI455X (gfx1250) — matrix operand order repaired and checked by reading, not compiled, not run on hardware
//
#include <hip/hip_runtime.h>
#include <hip/hip_bf16.h>

typedef __attribute__((ext_vector_type(16))) __bf16 v16bf;
typedef __attribute__((ext_vector_type(2)))  __bf16 v2bf;
typedef __attribute__((ext_vector_type(8)))  float  v8f;
typedef __attribute__((ext_vector_type(4)))  float  v4f;

#define DIMN 1024
#define NH   16
#define HD   64
#define SEQ  8192
#define NB   8
#define ASTR 36          // bf16 elements per A-slab row (32 + pad, even)
#define CSTR 132         // f32 elements per C row (128 + pad)
#define HALF_SIG2 0.03125f   // 0.5 * 0.25^2
#define SIGMA_C   0.25f

// native conversions: let clang emit v_cvt_*bf16* (RNE) instead of bit tricks
__device__ __forceinline__ unsigned short f2bf(float f) {
  union { __bf16 b; unsigned short u; } cv;
  cv.b = (__bf16)f;
  return cv.u;
}

__device__ __forceinline__ unsigned int pack_bf16x2(float lo, float hi) {
  union { v2bf b; unsigned int u; } cv;
  cv.b = (v2bf){(__bf16)lo, (__bf16)hi};
  return cv.u;
}

__device__ __forceinline__ v4f ldg_nt(const float* p) {
  return __builtin_nontemporal_load((const v4f*)p);
}

union FragB16 { v16bf v; unsigned int u[8]; };

// ---------------------------------------------------------------------------
// prep: bf16 transposed weights + zero accumulators
// ---------------------------------------------------------------------------
__global__ __launch_bounds__(256) void prep_kernel(
    const float* __restrict__ Wk, const float* __restrict__ Wv,
    const float* __restrict__ pp1w,
    unsigned short* __restrict__ WtK, unsigned short* __restrict__ WtV,
    unsigned short* __restrict__ pp1t,
    float* __restrict__ ctx_acc, float* __restrict__ wsum_acc) {
  int gid = blockIdx.x * 256 + threadIdx.x;
  if (gid < DIMN * DIMN) {
    int n = gid >> 10, k = gid & 1023;       // dest = [n][k]
    WtK[gid] = f2bf(Wk[k * DIMN + n]);
    WtV[gid] = f2bf(Wv[k * DIMN + n]);
  }
  if (gid < HD * HD) {
    int e = gid >> 6, d = gid & 63;          // dest = [e][d]
    pp1t[gid] = f2bf(pp1w[d * HD + e]);
  }
  if (gid < NB * DIMN) ctx_acc[gid] = 0.0f;
  if (gid < NB * NH)   wsum_acc[gid] = 0.0f;
}

// ---------------------------------------------------------------------------
// Qp = Q @ Wq + bq   (tiny, VALU)
// ---------------------------------------------------------------------------
__global__ __launch_bounds__(256) void qp_kernel(
    const float* __restrict__ Q, const float* __restrict__ Wq,
    const float* __restrict__ bq, float* __restrict__ Qp) {
  int gid = blockIdx.x * 256 + threadIdx.x;  // 8192
  int b = gid >> 10, n = gid & 1023;
  float acc = bq[n];
  const float* q = Q + b * DIMN;
#pragma unroll 4
  for (int k = 0; k < DIMN; ++k) acc = fmaf(q[k], Wq[k * DIMN + n], acc);
  Qp[gid] = acc;
}

// ---------------------------------------------------------------------------
// pipelined staging helper: pack 4 rows x 4 cols of fp32 regs into bf16 LDS
// ---------------------------------------------------------------------------
__device__ __forceinline__ void stash_slab(const v4f reg[4],
                                           unsigned short* Abuf,
                                           int srow, int scol) {
#pragma unroll
  for (int c = 0; c < 4; ++c) {
    unsigned int* dst = (unsigned int*)(Abuf + (srow + 32 * c) * ASTR + scol);
    dst[0] = pack_bf16x2(reg[c][0], reg[c][1]);
    dst[1] = pack_bf16x2(reg[c][2], reg[c][3]);
  }
}

// ---------------------------------------------------------------------------
// Core WMMA GEMM: C_lds[128 x 128] = X[l0..+128, :] @ Wt^T + bias[n0..+128]
// Wt is bf16, column-major ([n][k]).  8 waves, wave tile 64x32, K-slab 32,
// double-buffered A slab, streaming loads software-pipelined behind WMMA.
// ---------------------------------------------------------------------------
__device__ __forceinline__ void gemm128x128_bf16(
    const float* __restrict__ X, const unsigned short* __restrict__ Wt,
    const float* __restrict__ bias, int l0, int n0,
    unsigned short* A0, unsigned short* A1, float* C_lds, float* bias_s) {
  const int tid  = threadIdx.x;
  const int lane = tid & 31, wave = tid >> 5;
  const int wm = wave >> 2, wn = wave & 3;   // 2 (M) x 4 (N) waves
  const int g = lane >> 4, idx = lane & 15;
  const int srow = tid >> 3;                 // staging: 32 rows per chunk
  const int scol = (tid & 7) << 2;           // 4 contiguous cols

  if (tid < 128) bias_s[tid] = bias[n0 + tid];

  const float* xbase = X + (size_t)(l0 + srow) * DIMN + scol;

  // prologue: slab 0 -> regs -> A0
  v4f reg[4];
#pragma unroll
  for (int c = 0; c < 4; ++c) reg[c] = ldg_nt(xbase + (size_t)(c * 32) * DIMN);
  stash_slab(reg, A0, srow, scol);

  const v8f vzero = {0.f, 0.f, 0.f, 0.f, 0.f, 0.f, 0.f, 0.f};
  v8f acc[4][2];
#pragma unroll
  for (int i = 0; i < 4; ++i)
#pragma unroll
    for (int j = 0; j < 2; ++j) acc[i][j] = vzero;

  __syncthreads();

  for (int kt = 0; kt < 32; ++kt) {
    const unsigned short* Ac = (kt & 1) ? A1 : A0;
    unsigned short* An = (kt & 1) ? A0 : A1;
    const int kk = kt * 32;

    // prefetch next slab (nontemporal: K/V are streamed exactly once)
    if (kt < 31) {
#pragma unroll
      for (int c = 0; c < 4; ++c)
        reg[c] = ldg_nt(xbase + (size_t)(c * 32) * DIMN + kk + 32);
    }

    // B fragments (K x N): interleaved half-lane k-offset 8 per ISA table
    FragB16 fb[2];
#pragma unroll
    for (int ns = 0; ns < 2; ++ns) {
      const unsigned short* bp =
          Wt + (size_t)(n0 + wn * 32 + ns * 16 + idx) * DIMN + kk + g * 8;
#pragma unroll
      for (int i = 0; i < 8; ++i) fb[ns].u[i] = *(const unsigned int*)(bp + 2 * i + (i >= 4 ? 8 : 0));
    }
#pragma unroll
    for (int ms = 0; ms < 4; ++ms) {
      // A fragment (M x K): interleaved half-lane k-offset 8 per ISA table
      FragB16 fa;
      const unsigned short* ap = Ac + (wm * 64 + ms * 16 + idx) * ASTR + g * 8;
#pragma unroll
      for (int i = 0; i < 4; ++i) fa.u[i]     = *(const unsigned int*)(ap + 2 * i);
#pragma unroll
      for (int i = 0; i < 4; ++i) fa.u[4 + i] = *(const unsigned int*)(ap + 16 + 2 * i);
#pragma unroll
      for (int ns = 0; ns < 2; ++ns)
        acc[ms][ns] = __builtin_amdgcn_wmma_f32_16x16x32_bf16(
            false, fa.v, false, fb[ns].v, (short)0, acc[ms][ns], false, false);
    }

    if (kt < 31) stash_slab(reg, An, srow, scol);  // other buffer: no WAR here
    __syncthreads();                               // publish An for kt+1
  }

  // C/D layout: VGPR r -> row r (lanes 0-15) / r+8 (lanes 16-31), col = lane&15
#pragma unroll
  for (int ms = 0; ms < 4; ++ms)
#pragma unroll
    for (int ns = 0; ns < 2; ++ns)
#pragma unroll
      for (int r = 0; r < 8; ++r) {
        int row = wm * 64 + ms * 16 + g * 8 + r;
        int col = wn * 32 + ns * 16 + idx;
        C_lds[row * CSTR + col] = acc[ms][ns][r] + bias_s[col];
      }
  __syncthreads();
}

// ---------------------------------------------------------------------------
// K-projection + phi/num/wsum + prior-branch MLP (WMMA) + LN -> prior_mu
// grid: (SEQ/128, DIMN/128, NB)
// ---------------------------------------------------------------------------
__global__ __launch_bounds__(256) void kproj_kernel(
    const float* __restrict__ K, const unsigned short* __restrict__ WtK,
    const float* __restrict__ bk, const float* __restrict__ Qp,
    const unsigned short* __restrict__ pp1t, const float* __restrict__ pp1b,
    const float* __restrict__ ppg, const float* __restrict__ ppb,
    const float* __restrict__ pp2w, const float* __restrict__ pp2b,
    const unsigned char* __restrict__ padding, const float* __restrict__ eps,
    float* __restrict__ ws_num, float* __restrict__ wsum_acc,
    float* __restrict__ prior_out) {
  __shared__ unsigned short A_lds0[128 * ASTR];
  __shared__ unsigned short A_lds1[128 * ASTR];
  __shared__ float C_lds[128 * CSTR];
  __shared__ float bias_s[128];
  __shared__ float qp_s[128];
  __shared__ float pp1b_s[64], ppg_s[64], ppb_s[64], pp2_s[64];

  const int b = blockIdx.z;
  const int l0 = blockIdx.x * 128;
  const int n0 = blockIdx.y * 128;
  const int tid = threadIdx.x;

  if (tid < 128) qp_s[tid] = Qp[b * DIMN + n0 + tid];
  if (tid < 64) {
    pp1b_s[tid] = pp1b[tid]; ppg_s[tid] = ppg[tid];
    ppb_s[tid] = ppb[tid];   pp2_s[tid] = pp2w[tid];
  }

  gemm128x128_bf16(K + (size_t)b * SEQ * DIMN, WtK, bk, l0, n0,
                   A_lds0, A_lds1, C_lds, bias_s);

  // ---- phi / posterior sample / num / wsum --------------------------------
  const int h0 = n0 >> 6;
  const int l_local = tid & 127, hl = tid >> 7;
  const int h = h0 + hl, l = l0 + l_local;
  {
    const float* kv = C_lds + l_local * CSTR + hl * 64;
    float phi = 0.0f;
#pragma unroll 8
    for (int d = 0; d < 64; ++d) phi = fmaf(qp_s[hl * 64 + d], kv[d], phi);
    phi *= 0.125f;                                   // / sqrt(HD)
    float num = 0.0f;
    if (!padding[b * SEQ + l]) {
      float e = eps[(size_t)(b * NH + h) * SEQ + l];
      num = __expf(2.0f * ((phi - HALF_SIG2) + SIGMA_C * e));  // exp(s)^2
    }
    ws_num[(size_t)(b * NH + h) * SEQ + l] = num;
    float s = num;                                   // wave reduce (all same h)
#pragma unroll
    for (int off = 16; off > 0; off >>= 1) s += __shfl_xor(s, off);
    if ((tid & 31) == 0) atomicAdd(&wsum_acc[b * NH + h], s);
  }
  __syncthreads();

  // ---- prior MLP: per head  H1 = Kp[128x64] @ pp1[64x64]  (WMMA) ----------
  {
    const int lane = tid & 31, wave = tid >> 5;
    const int g = lane >> 4, idx = lane & 15;
    const int hl2 = wave >> 2, ns = wave & 3;        // 2 heads x 4 N-subtiles
    const v8f vzero = {0.f, 0.f, 0.f, 0.f, 0.f, 0.f, 0.f, 0.f};
    v8f pacc[8];
#pragma unroll
    for (int ms = 0; ms < 8; ++ms) {
      v8f a = vzero;
#pragma unroll
      for (int ks = 0; ks < 2; ++ks) {
        FragB16 fa;
        const float* ap = C_lds + (ms * 16 + idx) * CSTR + hl2 * 64 + ks * 32;
#pragma unroll
        for (int i = 0; i < 4; ++i) {
          int k0 = g * 8 + 2 * i;
          fa.u[i] = pack_bf16x2(ap[k0], ap[k0 + 1]);
        }
#pragma unroll
        for (int i = 0; i < 4; ++i) {
          int k0 = 16 + g * 8 + 2 * i;
          fa.u[4 + i] = pack_bf16x2(ap[k0], ap[k0 + 1]);
        }
        FragB16 fbp;
        const unsigned short* bp = pp1t + (ns * 16 + idx) * HD + ks * 32 + g * 8;
#pragma unroll
        for (int i = 0; i < 8; ++i) fbp.u[i] = *(const unsigned int*)(bp + 2 * i + (i >= 4 ? 8 : 0));
        a = __builtin_amdgcn_wmma_f32_16x16x32_bf16(
            false, fa.v, false, fbp.v, (short)0, a, false, false);
      }
      pacc[ms] = a;
    }
    __syncthreads();       // all Kp reads done before overwrite
#pragma unroll
    for (int ms = 0; ms < 8; ++ms)
#pragma unroll
      for (int r = 0; r < 8; ++r)
        C_lds[(ms * 16 + g * 8 + r) * CSTR + hl2 * 64 + ns * 16 + idx] =
            pacc[ms][r] + pp1b_s[ns * 16 + idx];
  }
  __syncthreads();

  // ---- LN + relu + dot(pp2) -> prior_mu -----------------------------------
  {
    const float* hv = C_lds + l_local * CSTR + hl * 64;
    float m = 0.0f;
#pragma unroll 8
    for (int d = 0; d < 64; ++d) m += hv[d];
    m *= (1.0f / 64.0f);
    float v = 0.0f;
#pragma unroll 8
    for (int d = 0; d < 64; ++d) { float t = hv[d] - m; v = fmaf(t, t, v); }
    v *= (1.0f / 64.0f);
    float inv = rsqrtf(v + 1e-5f);
    float pf = 0.0f;
#pragma unroll 8
    for (int d = 0; d < 64; ++d) {
      float x = (hv[d] - m) * inv * ppg_s[d] + ppb_s[d];
      pf = fmaf(fmaxf(x, 0.0f), pp2_s[d], pf);
    }
    prior_out[(size_t)(b * NH + h) * SEQ + l] = pf + pp2b[0] - HALF_SIG2;
  }
}

// ---------------------------------------------------------------------------
// V-projection + unnormalized ctx accumulation
// ---------------------------------------------------------------------------
__global__ __launch_bounds__(256) void vproj_kernel(
    const float* __restrict__ V, const unsigned short* __restrict__ WtV,
    const float* __restrict__ bv, const float* __restrict__ ws_num,
    float* __restrict__ ctx_acc) {
  __shared__ unsigned short A_lds0[128 * ASTR];
  __shared__ unsigned short A_lds1[128 * ASTR];
  __shared__ float C_lds[128 * CSTR];
  __shared__ float bias_s[128];
  __shared__ float num_lds[256];

  const int b = blockIdx.z;
  const int l0 = blockIdx.x * 128;
  const int n0 = blockIdx.y * 128;
  const int tid = threadIdx.x;

  gemm128x128_bf16(V + (size_t)b * SEQ * DIMN, WtV, bv, l0, n0,
                   A_lds0, A_lds1, C_lds, bias_s);

  const int h0 = n0 >> 6;
  {
    int hl = tid >> 7, l = tid & 127;
    num_lds[hl * 128 + l] = ws_num[(size_t)(b * NH + h0 + hl) * SEQ + l0 + l];
  }
  __syncthreads();

  const int n_local = tid & 127, half = tid >> 7;
  const float* col = C_lds + (half * 64) * CSTR + n_local;
  const float* nm  = num_lds + (n_local >> 6) * 128 + half * 64;
  float s = 0.0f;
#pragma unroll 8
  for (int i = 0; i < 64; ++i) s = fmaf(nm[i], col[i * CSTR], s);
  atomicAdd(&ctx_acc[(b * NH + h0 + (n_local >> 6)) * HD + (n_local & 63)], s);
}

// ---------------------------------------------------------------------------
// final: ctx scale, ctx@Wo+bo, LN, +Q   (one block per batch row)
// ---------------------------------------------------------------------------
__global__ __launch_bounds__(256) void out_kernel(
    const float* __restrict__ Q, const float* __restrict__ Wo,
    const float* __restrict__ bo, const float* __restrict__ ln_g,
    const float* __restrict__ ln_b, const float* __restrict__ ctx_acc,
    const float* __restrict__ wsum_acc, float* __restrict__ out) {
  __shared__ float ctx_s[DIMN];
  __shared__ float red[256];
  const int b = blockIdx.x, tid = threadIdx.x;

  for (int i = tid; i < DIMN; i += 256) {
    float w = wsum_acc[b * NH + (i >> 6)] + 1e-8f;
    ctx_s[i] = ctx_acc[b * DIMN + i] * __powf(w, -0.25f);
  }
  __syncthreads();

  float y[4];
#pragma unroll
  for (int jj = 0; jj < 4; ++jj) y[jj] = bo[tid + jj * 256];
  for (int d = 0; d < DIMN; ++d) {
    float c = ctx_s[d];
    const float* wrow = Wo + (size_t)d * DIMN;
#pragma unroll
    for (int jj = 0; jj < 4; ++jj) y[jj] = fmaf(c, wrow[tid + jj * 256], y[jj]);
  }
  float ps = y[0] + y[1] + y[2] + y[3];
  red[tid] = ps; __syncthreads();
  for (int off = 128; off > 0; off >>= 1) {
    if (tid < off) red[tid] += red[tid + off];
    __syncthreads();
  }
  float mean = red[0] * (1.0f / DIMN);
  __syncthreads();
  float pv = 0.0f;
#pragma unroll
  for (int jj = 0; jj < 4; ++jj) { float t = y[jj] - mean; pv = fmaf(t, t, pv); }
  red[tid] = pv; __syncthreads();
  for (int off = 128; off > 0; off >>= 1) {
    if (tid < off) red[tid] += red[tid + off];
    __syncthreads();
  }
  float var = red[0] * (1.0f / DIMN);
  float inv = rsqrtf(var + 1e-5f);
#pragma unroll
  for (int jj = 0; jj < 4; ++jj) {
    int j = tid + jj * 256;
    out[b * DIMN + j] = (y[jj] - mean) * inv * ln_g[j] + ln_b[j] + Q[b * DIMN + j];
  }
}

// ---------------------------------------------------------------------------
extern "C" void kernel_launch(void* const* d_in, const int* in_sizes, int n_in,
                              void* d_out, int out_size, void* d_ws, size_t ws_size,
                              hipStream_t stream) {
  const float* Q   = (const float*)d_in[0];
  const float* K   = (const float*)d_in[1];
  const float* V   = (const float*)d_in[2];
  const unsigned char* padding = (const unsigned char*)d_in[3];
  const float* eps = (const float*)d_in[4];
  const float* Wq  = (const float*)d_in[5];
  const float* bq  = (const float*)d_in[6];
  const float* Wk  = (const float*)d_in[7];
  const float* bk  = (const float*)d_in[8];
  const float* Wv  = (const float*)d_in[9];
  const float* bv  = (const float*)d_in[10];
  const float* Wo  = (const float*)d_in[11];
  const float* bo  = (const float*)d_in[12];
  const float* ln_g = (const float*)d_in[13];
  const float* ln_b = (const float*)d_in[14];
  const float* pp1w = (const float*)d_in[15];
  const float* pp1b = (const float*)d_in[16];
  const float* ppg  = (const float*)d_in[17];
  const float* ppb  = (const float*)d_in[18];
  const float* pp2w = (const float*)d_in[19];
  const float* pp2b = (const float*)d_in[20];

  char* ws = (char*)d_ws;
  unsigned short* WtK  = (unsigned short*)(ws + 0);
  unsigned short* WtV  = (unsigned short*)(ws + 2097152);
  unsigned short* pp1t = (unsigned short*)(ws + 4194304);
  float* Qp     = (float*)(ws + 4202496);
  float* ws_num = (float*)(ws + 4235264);
  float* wsum   = (float*)(ws + 8429568);
  float* ctx    = (float*)(ws + 8430080);

  float* out   = (float*)d_out;
  float* prior = out + NB * DIMN;

  prep_kernel<<<4096, 256, 0, stream>>>(Wk, Wv, pp1w, WtK, WtV, pp1t, ctx, wsum);
  qp_kernel<<<32, 256, 0, stream>>>(Q, Wq, bq, Qp);

  dim3 grid(SEQ / 128, DIMN / 128, NB);  // (64, 8, 8)
  kproj_kernel<<<grid, 256, 0, stream>>>(K, WtK, bk, Qp, pp1t, pp1b, ppg, ppb,
                                         pp2w, pp2b, padding, eps,
                                         ws_num, wsum, prior);
  vproj_kernel<<<grid, 256, 0, stream>>>(V, WtV, bv, ws_num, ctx);
  out_kernel<<<NB, 256, 0, stream>>>(Q, Wo, bo, ln_g, ln_b, ctx, wsum, out);
}
